// Transformer_63161789055026
// MI455X (gfx1250) — compile-verified, matrix operand order checked by reading, not run on hardware
//
#include <hip/hip_runtime.h>
#include <hip/hip_bf16.h>
#include <cstdint>

// ---------------------------------------------------------------------------
// MI455X (gfx1250) transformer forward. Compute-bound (~2000 FLOP/byte) =>
// all GEMMs through v_wmma_f32_16x16x32_f16 with fp32 accumulation.
// Weights are converted fp32->f16 *and transposed* once per launch so every
// global->LDS tile copy is a contiguous B128; GEMM staging uses the CDNA5
// async global->LDS path (ASYNCcnt). Workspace use: ~150 MB.
// ---------------------------------------------------------------------------

#define DEPTH    4
#define DIM      768
#define HEADS    12
#define DH       64
#define INNER    768
#define FF_INNER 2048
#define BATCH    4
#define SEQ      1024
#define NTOK     (BATCH * SEQ)     // 4096 token rows
#define ATTN_SCALE 8.0f

typedef __attribute__((ext_vector_type(16))) _Float16 v16h;
typedef __attribute__((ext_vector_type(8)))  _Float16 v8h;
typedef __attribute__((ext_vector_type(8)))  float    v8f;

#define WMMA_F32_F16(A, B, C) \
  __builtin_amdgcn_wmma_f32_16x16x32_f16(false, (A), false, (B), (short)0, (C), false, false)

// ---- CDNA5 async global->LDS (ISA 15.18.3 opc 98) -------------------------
// VDST = per-lane LDS byte address, VADDR = per-lane 64-bit global address.
// On AMDGPU the low 32 bits of a flat shared pointer are the LDS offset.
__device__ __forceinline__ void async_load_b128(const _Float16* lds_dst,
                                                const _Float16* gsrc) {
  uint32_t lds_addr = (uint32_t)(uintptr_t)lds_dst;
  uint64_t gaddr = (uint64_t)(uintptr_t)gsrc;
  asm volatile("global_load_async_to_lds_b128 %0, %1, off"
               :: "v"(lds_addr), "v"(gaddr)
               : "memory");
}
__device__ __forceinline__ void wait_async0() {
  asm volatile("s_wait_asynccnt 0" ::: "memory");
}

// ---- WMMA fragment loaders (wave32 layouts per CDNA5 ISA 7.12.2) ----------
// A 16x32 f16: lanes 0-15 hold row M=lane with K in {0..7,16..23};
//              lanes 16-31 hold row M=lane-16 with K in {8..15,24..31}.
// B fragments use the same pattern on an [N][K]-stored tile.
__device__ __forceinline__ v16h frag_from_h16(const _Float16* __restrict__ base,
                                              int stride, int lane) {
  const int m  = lane & 15;
  const int kb = (lane < 16) ? 0 : 8;
  const _Float16* p = base + m * stride + kb;
  v8h lo = *(const v8h*)(p);
  v8h hi = *(const v8h*)(p + 16);
  v16h r;
#pragma unroll
  for (int i = 0; i < 8; ++i) { r[i] = lo[i]; r[i + 8] = hi[i]; }
  return r;
}

__device__ __forceinline__ v16h frag_from_f32(const float* __restrict__ base,
                                              int stride, int lane) {
  const int m  = lane & 15;
  const int kb = (lane < 16) ? 0 : 8;
  const float* p = base + m * stride + kb;
  v16h r;
#pragma unroll
  for (int i = 0; i < 8; ++i) {
    r[i]     = (_Float16)p[i];
    r[i + 8] = (_Float16)p[i + 16];
  }
  return r;
}

// ---------------------------------------------------------------------------
// Elementwise utility kernels
// ---------------------------------------------------------------------------
__global__ void cast_f32_to_f16(const float* __restrict__ src,
                                _Float16* __restrict__ dst, size_t n) {
  size_t i = (size_t)blockIdx.x * blockDim.x + threadIdx.x;
  size_t stride = (size_t)gridDim.x * blockDim.x;
  for (; i < n; i += stride) dst[i] = (_Float16)src[i];
}

// Weight convert + transpose: dst[l][n][k] = (f16) src[l][k][n].
// Output-coalesced; one-time cost amortized over the whole forward pass.
__global__ void cast_transpose_f16(const float* __restrict__ src,
                                   _Float16* __restrict__ dst,
                                   int K, int N, size_t total) {
  size_t j = (size_t)blockIdx.x * blockDim.x + threadIdx.x;
  size_t stride = (size_t)gridDim.x * blockDim.x;
  size_t plane = (size_t)K * N;
  for (; j < total; j += stride) {
    int k = (int)(j % K);
    size_t t = j / K;
    int n = (int)(t % N);
    size_t l = t / N;
    dst[j] = (_Float16)src[l * plane + (size_t)k * N + n];
  }
}

__global__ void copy_f32(const float* __restrict__ src, float* __restrict__ dst,
                         size_t n) {
  size_t i = (size_t)blockIdx.x * blockDim.x + threadIdx.x;
  size_t stride = (size_t)gridDim.x * blockDim.x;
  for (; i < n; i += stride) dst[i] = src[i];
}

// LayerNorm over DIM=768, one block per row, f16 output for WMMA consumption.
__global__ __launch_bounds__(256)
void ln_to_f16(const float* __restrict__ x, const float* __restrict__ gamma,
               const float* __restrict__ beta, _Float16* __restrict__ out) {
  __shared__ float red[256];
  const int tid = threadIdx.x;
  const size_t row = blockIdx.x;
  const float* xr = x + row * DIM;
  float v0 = xr[tid], v1 = xr[tid + 256], v2 = xr[tid + 512];
  red[tid] = v0 + v1 + v2;
  __syncthreads();
  for (int s = 128; s > 0; s >>= 1) {
    if (tid < s) red[tid] += red[tid + s];
    __syncthreads();
  }
  const float mu = red[0] * (1.0f / DIM);
  __syncthreads();
  float d0 = v0 - mu, d1 = v1 - mu, d2 = v2 - mu;
  red[tid] = d0 * d0 + d1 * d1 + d2 * d2;
  __syncthreads();
  for (int s = 128; s > 0; s >>= 1) {
    if (tid < s) red[tid] += red[tid + s];
    __syncthreads();
  }
  const float rs = rsqrtf(red[0] * (1.0f / DIM) + 1e-5f);
  const float b0 = beta ? beta[tid] : 0.f;
  const float b1 = beta ? beta[tid + 256] : 0.f;
  const float b2 = beta ? beta[tid + 512] : 0.f;
  _Float16* o = out + row * DIM;
  o[tid]       = (_Float16)(d0 * rs * gamma[tid] + b0);
  o[tid + 256] = (_Float16)(d1 * rs * gamma[tid + 256] + b1);
  o[tid + 512] = (_Float16)(d2 * rs * gamma[tid + 512] + b2);
}

// RoPE (2D axial, H=32) + L2-normalize + per-dim scale; writes [B,H,N,DH] f16.
__global__ __launch_bounds__(64)
void rope_l2norm_f16(const float* __restrict__ src, int src_stride, int src_off,
                     const float* __restrict__ scale, _Float16* __restrict__ dst) {
  __shared__ float buf[DH];
  __shared__ float red[DH];
  const int idx = blockIdx.x;              // (b*SEQ + n)*HEADS + h
  const int h = idx % HEADS;
  const int bn = idx / HEADS;
  const int n = bn % SEQ;
  const int b = bn / SEQ;
  const float* t = src + (size_t)bn * src_stride + src_off + h * DH;
  const int d = threadIdx.x;
  if (d < DH / 2) {
    // pair p=d: freq index p/2, x-angle if p even else y-angle (H = 32)
    float fr = __powf(10000.f, -(float)((d >> 1) * 4) * (1.0f / DH));
    float pv = (d & 1) ? (float)(n >> 5) : (float)(n & 31);
    float ang = pv * fr;
    float s, c;
    __sincosf(ang, &s, &c);
    float re = t[2 * d], im = t[2 * d + 1];
    buf[2 * d]     = re * c - im * s;
    buf[2 * d + 1] = re * s + im * c;
  }
  __syncthreads();
  float v = buf[d];
  red[d] = v * v;
  __syncthreads();
  for (int st = 32; st > 0; st >>= 1) {
    if (d < st) red[d] += red[d + st];
    __syncthreads();
  }
  float nrm = fmaxf(sqrtf(red[0]), 1e-12f);
  dst[(((size_t)b * HEADS + h) * SEQ + n) * DH + d] = (_Float16)(v / nrm * scale[d]);
}

// V half of KV -> head-transposed [B,H,DH,SEQ] f16 so flash V tiles load
// contiguously (no LDS scatter). Output-coalesced.
__global__ void v_transpose_f16(const float* __restrict__ kv,
                                _Float16* __restrict__ vht, size_t total) {
  size_t j = (size_t)blockIdx.x * blockDim.x + threadIdx.x;
  size_t stride = (size_t)gridDim.x * blockDim.x;
  for (; j < total; j += stride) {
    int n = (int)(j & (SEQ - 1));
    size_t t = j >> 10;
    int d = (int)(t & (DH - 1));
    size_t bh = t >> 6;
    int h = (int)(bh % HEADS);
    int b = (int)(bh / HEADS);
    float v = kv[((size_t)b * SEQ + n) * (2 * INNER) + INNER + h * DH + d];
    vht[j] = (_Float16)v;
  }
}

// ---------------------------------------------------------------------------
// Generic f16 x f16^T -> f32 GEMM. Block tile 128x64, 8 waves, each wave a
// 32x32 slab (2 A-frags x 2 B-frags -> 4 WMMA per 32-wide K step). B operand
// is pre-transposed (Bt[n][k]) so both stagings are contiguous B128 async
// global->LDS copies. RESIDUAL=true fuses C += acc.
// ---------------------------------------------------------------------------
template <bool RESIDUAL>
__global__ __launch_bounds__(256)
void gemm_f16_f32(const _Float16* __restrict__ A, int lda,
                  const _Float16* __restrict__ Bt, int ldbt,
                  float* __restrict__ C, int ldc, int K) {
  __shared__ __align__(16) _Float16 Als[128][32];
  __shared__ __align__(16) _Float16 Bls[64][32];   // [n][k]
  const int tid = threadIdx.x, lane = tid & 31, wave = tid >> 5;
  const int row0 = blockIdx.x * 128, col0 = blockIdx.y * 64;
  const int wm = (wave & 3) * 32, wn = (wave >> 2) * 32;
  // A tile: 128x32 halves = 512 x (8-half segs); 2 segs per thread.
  const int as0 = tid * 2, as1 = tid * 2 + 1;
  const int ar0 = as0 >> 2, ao0 = (as0 & 3) * 8;
  const int ar1 = as1 >> 2, ao1 = (as1 & 3) * 8;
  // B tile: 64x32 halves = 256 segs; 1 per thread.
  const int bn = tid >> 2, bo = (tid & 3) * 8;
  v8f c00 = {}, c01 = {}, c10 = {}, c11 = {};
  for (int k0 = 0; k0 < K; k0 += 32) {
    async_load_b128(&Als[ar0][ao0], A + (size_t)(row0 + ar0) * lda + k0 + ao0);
    async_load_b128(&Als[ar1][ao1], A + (size_t)(row0 + ar1) * lda + k0 + ao1);
    async_load_b128(&Bls[bn][bo],   Bt + (size_t)(col0 + bn) * ldbt + k0 + bo);
    wait_async0();
    __syncthreads();
    v16h a0 = frag_from_h16(&Als[wm][0], 32, lane);
    v16h a1 = frag_from_h16(&Als[wm + 16][0], 32, lane);
    v16h b0 = frag_from_h16(&Bls[wn][0], 32, lane);
    v16h b1 = frag_from_h16(&Bls[wn + 16][0], 32, lane);
    c00 = WMMA_F32_F16(a0, b0, c00);
    c01 = WMMA_F32_F16(a0, b1, c01);
    c10 = WMMA_F32_F16(a1, b0, c10);
    c11 = WMMA_F32_F16(a1, b1, c11);
    __syncthreads();
  }
  const int n = lane & 15, moff = (lane >> 4) * 8;
#pragma unroll
  for (int r = 0; r < 8; ++r) {
    size_t i00 = (size_t)(row0 + wm + moff + r) * ldc + col0 + wn + n;
    size_t i10 = i00 + (size_t)16 * ldc;
    if (RESIDUAL) {
      C[i00] += c00[r]; C[i00 + 16] += c01[r];
      C[i10] += c10[r]; C[i10 + 16] += c11[r];
    } else {
      C[i00] = c00[r]; C[i00 + 16] = c01[r];
      C[i10] = c10[r]; C[i10 + 16] = c11[r];
    }
  }
}

// ---------------------------------------------------------------------------
// FF1 with fused SwiGLU: block computes matching a-tile (cols col0) and
// g-tile (cols col0+FF_INNER) of h @ wff1 (W1t pre-transposed [4096][768]),
// writes a * gelu_exact(g) as f16. 64x64 block, wave = 16x32 slab.
// ---------------------------------------------------------------------------
__global__ __launch_bounds__(256)
void gemm_ff1_swiglu(const _Float16* __restrict__ A, const _Float16* __restrict__ W1t,
                     _Float16* __restrict__ act) {
  __shared__ __align__(16) _Float16 Als[64][32];
  __shared__ __align__(16) _Float16 Bas[64][32];
  __shared__ __align__(16) _Float16 Bgs[64][32];
  const int tid = threadIdx.x, lane = tid & 31, wave = tid >> 5;
  const int row0 = blockIdx.x * 64, col0 = blockIdx.y * 64;
  const int wm = (wave & 3) * 16, wn = (wave >> 2) * 32;
  const int ar = tid >> 2, as = (tid & 3) * 8;
  const int bn = tid >> 2, bo = (tid & 3) * 8;
  v8f a0 = {}, a1 = {}, g0 = {}, g1 = {};
  for (int k0 = 0; k0 < DIM; k0 += 32) {
    *(v8h*)&Als[ar][as] = *(const v8h*)(A + (size_t)(row0 + ar) * DIM + k0 + as);
    *(v8h*)&Bas[bn][bo] =
        *(const v8h*)(W1t + (size_t)(col0 + bn) * DIM + k0 + bo);
    *(v8h*)&Bgs[bn][bo] =
        *(const v8h*)(W1t + (size_t)(FF_INNER + col0 + bn) * DIM + k0 + bo);
    if (k0 + 32 < DIM)  // stream next weight tile through gfx1250 prefetch path
      __builtin_prefetch(W1t + (size_t)(col0 + bn) * DIM + k0 + 32 + bo, 0, 1);
    __syncthreads();
    v16h af  = frag_from_h16(&Als[wm][0], 32, lane);
    v16h ba0 = frag_from_h16(&Bas[wn][0], 32, lane);
    v16h ba1 = frag_from_h16(&Bas[wn + 16][0], 32, lane);
    v16h bg0 = frag_from_h16(&Bgs[wn][0], 32, lane);
    v16h bg1 = frag_from_h16(&Bgs[wn + 16][0], 32, lane);
    a0 = WMMA_F32_F16(af, ba0, a0);
    a1 = WMMA_F32_F16(af, ba1, a1);
    g0 = WMMA_F32_F16(af, bg0, g0);
    g1 = WMMA_F32_F16(af, bg1, g1);
    __syncthreads();
  }
  const int n = lane & 15, moff = (lane >> 4) * 8;
#pragma unroll
  for (int r = 0; r < 8; ++r) {
    size_t base = (size_t)(row0 + wm + moff + r) * FF_INNER + col0 + wn + n;
    float ga = g0[r], gb = g1[r];
    float e0 = 0.5f * ga * (1.0f + erff(ga * 0.70710678f));
    float e1 = 0.5f * gb * (1.0f + erff(gb * 0.70710678f));
    act[base]      = (_Float16)(a0[r] * e0);
    act[base + 16] = (_Float16)(a1[r] * e1);
  }
}

// ---------------------------------------------------------------------------
// FlashAttention-2: block = (b,h, 64-row Q tile), 8 waves. Streams 64-wide
// K/V tiles; S tile and P*V via WMMA; online softmax with O rescale.
// V comes pre-transposed [B,H,DH,SEQ] so all tile loads are contiguous.
// ---------------------------------------------------------------------------
__global__ __launch_bounds__(256)
void flash_attn_kernel(const _Float16* __restrict__ qh, const _Float16* __restrict__ kh,
                       const _Float16* __restrict__ vht, _Float16* __restrict__ o16) {
  __shared__ __align__(16) _Float16 Qls[64 * 64];
  __shared__ __align__(16) _Float16 Kls[64 * 64];
  __shared__ __align__(16) _Float16 Vls[64 * 64];  // [d][pos]
  __shared__ __align__(16) float    Sls[64 * 64];
  __shared__ float rowM[64], rowL[64], rowC[64];
  const int tid = threadIdx.x, lane = tid & 31, wave = tid >> 5;
  const int bh = blockIdx.x, rt = blockIdx.y;
  const int b = bh / HEADS, h = bh % HEADS;
  const _Float16* Qb = qh + ((size_t)bh * SEQ + (size_t)rt * 64) * DH;
  const _Float16* Kb = kh + (size_t)bh * SEQ * DH;
  const _Float16* Vb = vht + (size_t)bh * DH * SEQ;   // [d][n]
  const int lr = tid >> 2, ls = (tid & 3) * 16;
  *(v8h*)&Qls[lr * 64 + ls]     = *(const v8h*)(Qb + lr * 64 + ls);
  *(v8h*)&Qls[lr * 64 + ls + 8] = *(const v8h*)(Qb + lr * 64 + ls + 8);
  if (tid < 64) { rowM[tid] = -3.4e38f; rowL[tid] = 0.f; }
  const int wm = (wave & 3) * 16, wn = (wave >> 2) * 32;
  const int n = lane & 15, moff = (lane >> 4) * 8;
  v8f o0 = {}, o1 = {};
  for (int kt = 0; kt < SEQ / 64; ++kt) {
    __syncthreads();  // prior iteration's P/V reads done
    {
      const _Float16* kp = Kb + (size_t)(kt * 64 + lr) * 64 + ls;   // K[j][d]
      *(v8h*)&Kls[lr * 64 + ls]     = *(const v8h*)kp;
      *(v8h*)&Kls[lr * 64 + ls + 8] = *(const v8h*)(kp + 8);
      const _Float16* vp = Vb + (size_t)lr * SEQ + kt * 64 + ls;    // V^T[d][n]
      *(v8h*)&Vls[lr * 64 + ls]     = *(const v8h*)vp;
      *(v8h*)&Vls[lr * 64 + ls + 8] = *(const v8h*)(vp + 8);
    }
    __syncthreads();
    // S = Q K^T * scale
    v8f s0 = {}, s1 = {};
#pragma unroll
    for (int kb = 0; kb < DH; kb += 32) {
      v16h aq  = frag_from_h16(&Qls[wm * 64 + kb], 64, lane);
      v16h k0f = frag_from_h16(&Kls[wn * 64 + kb], 64, lane);
      v16h k1f = frag_from_h16(&Kls[(wn + 16) * 64 + kb], 64, lane);
      s0 = WMMA_F32_F16(aq, k0f, s0);
      s1 = WMMA_F32_F16(aq, k1f, s1);
    }
#pragma unroll
    for (int r = 0; r < 8; ++r) {
      Sls[(wm + moff + r) * 64 + wn + n]      = s0[r] * ATTN_SCALE;
      Sls[(wm + moff + r) * 64 + wn + 16 + n] = s1[r] * ATTN_SCALE;
    }
    __syncthreads();
    // online softmax update (one thread per row)
    if (tid < 64) {
      float* row = &Sls[tid * 64];
      float m0 = rowM[tid], mx = m0;
#pragma unroll 4
      for (int c = 0; c < 64; ++c) mx = fmaxf(mx, row[c]);
      float sum = 0.f;
#pragma unroll 4
      for (int c = 0; c < 64; ++c) { float e = __expf(row[c] - mx); row[c] = e; sum += e; }
      float corr = __expf(m0 - mx);
      rowC[tid] = corr;
      rowL[tid] = rowL[tid] * corr + sum;
      rowM[tid] = mx;
    }
    __syncthreads();
#pragma unroll
    for (int r = 0; r < 8; ++r) {
      float f = rowC[wm + moff + r];
      o0[r] *= f;
      o1[r] *= f;
    }
    // O += P V  (P fragments converted f32->f16 straight from LDS)
#pragma unroll
    for (int kb = 0; kb < 64; kb += 32) {
      v16h ap  = frag_from_f32(&Sls[wm * 64 + kb], 64, lane);
      v16h v0f = frag_from_h16(&Vls[wn * 64 + kb], 64, lane);
      v16h v1f = frag_from_h16(&Vls[(wn + 16) * 64 + kb], 64, lane);
      o0 = WMMA_F32_F16(ap, v0f, o0);
      o1 = WMMA_F32_F16(ap, v1f, o1);
    }
  }
  __syncthreads();
#pragma unroll
  for (int r = 0; r < 8; ++r) {
    int row = wm + moff + r;
    float inv = 1.0f / fmaxf(rowL[row], 1e-20f);
    size_t pos = (size_t)rt * 64 + row;
    size_t base = ((size_t)b * SEQ + pos) * INNER + (size_t)h * DH;
    o16[base + wn + n]      = (_Float16)(o0[r] * inv);
    o16[base + wn + 16 + n] = (_Float16)(o1[r] * inv);
  }
}

// ---------------------------------------------------------------------------
// Host orchestration
// ---------------------------------------------------------------------------
extern "C" void kernel_launch(void* const* d_in, const int* in_sizes, int n_in,
                              void* d_out, int out_size, void* d_ws, size_t ws_size,
                              hipStream_t stream) {
  (void)in_sizes; (void)n_in; (void)out_size; (void)ws_size;
  const float* x_in    = (const float*)d_in[0];
  const float* attn_g  = (const float*)d_in[1];
  const float* wq      = (const float*)d_in[2];
  const float* wkv     = (const float*)d_in[3];
  const float* q_scale = (const float*)d_in[4];
  const float* k_scale = (const float*)d_in[5];
  const float* wo      = (const float*)d_in[6];
  const float* ff_g    = (const float*)d_in[7];
  const float* ff_b    = (const float*)d_in[8];
  const float* wff1    = (const float*)d_in[9];
  const float* wff2    = (const float*)d_in[10];
  float* x = (float*)d_out;  // running residual stream lives in d_out

  char* ws = (char*)d_ws;
  size_t off = 0;
  auto take = [&](size_t bytes) -> char* {
    char* p = ws + off;
    off = (off + bytes + 255) & ~(size_t)255;
    return p;
  };
  // Transposed f16 weights: [l][n][k]
  _Float16* wq16   = (_Float16*)take((size_t)DEPTH * DIM * INNER * 2);
  _Float16* wkv16  = (_Float16*)take((size_t)DEPTH * DIM * 2 * INNER * 2);
  _Float16* wo16   = (_Float16*)take((size_t)DEPTH * INNER * DIM * 2);
  _Float16* wff116 = (_Float16*)take((size_t)DEPTH * DIM * 2 * FF_INNER * 2);
  _Float16* wff216 = (_Float16*)take((size_t)DEPTH * FF_INNER * DIM * 2);
  _Float16* h16    = (_Float16*)take((size_t)NTOK * DIM * 2);
  _Float16* x16    = (_Float16*)take((size_t)NTOK * DIM * 2);
  float*    qbuf   = (float*)take((size_t)NTOK * INNER * 4);
  float*    kvbuf  = (float*)take((size_t)NTOK * 2 * INNER * 4);
  _Float16* qh16   = (_Float16*)take((size_t)NTOK * INNER * 2);
  _Float16* kh16   = (_Float16*)take((size_t)NTOK * INNER * 2);
  _Float16* vht16  = (_Float16*)take((size_t)NTOK * INNER * 2);
  _Float16* o16    = (_Float16*)take((size_t)NTOK * INNER * 2);
  _Float16* act16  = (_Float16*)take((size_t)NTOK * FF_INNER * 2);

  auto tcast = [&](const float* s, _Float16* dst, int K, int N) {
    size_t cnt = (size_t)DEPTH * K * N;
    int blocks = (int)((cnt + 2047) / 2048);
    cast_transpose_f16<<<blocks, 256, 0, stream>>>(s, dst, K, N, cnt);
  };
  // fp32 -> f16 weight conversion + transpose (deterministic, every call)
  tcast(wq,   wq16,   DIM, INNER);
  tcast(wkv,  wkv16,  DIM, 2 * INNER);
  tcast(wo,   wo16,   INNER, DIM);
  tcast(wff1, wff116, DIM, 2 * FF_INNER);
  tcast(wff2, wff216, FF_INNER, DIM);

  copy_f32<<<1536, 256, 0, stream>>>(x_in, x, (size_t)NTOK * DIM);

  const dim3 gq(NTOK / 128, INNER / 64);
  const dim3 gkv(NTOK / 128, 2 * INNER / 64);
  const dim3 gattn(BATCH * HEADS, SEQ / 64);
  const dim3 gff1(NTOK / 64, FF_INNER / 64);
  const dim3 gff2(NTOK / 128, DIM / 64);

  for (int l = 0; l < DEPTH; ++l) {
    const _Float16* Wq  = wq16   + (size_t)l * DIM * INNER;
    const _Float16* Wkv = wkv16  + (size_t)l * DIM * 2 * INNER;
    const _Float16* Wo  = wo16   + (size_t)l * INNER * DIM;
    const _Float16* W1  = wff116 + (size_t)l * DIM * 2 * FF_INNER;
    const _Float16* W2  = wff216 + (size_t)l * FF_INNER * DIM;

    // attention branch
    ln_to_f16<<<NTOK, 256, 0, stream>>>(x, attn_g + (size_t)l * DIM, nullptr, h16);
    cast_f32_to_f16<<<1536, 256, 0, stream>>>(x, x16, (size_t)NTOK * DIM);
    gemm_f16_f32<false><<<gq, 256, 0, stream>>>(h16, DIM, Wq, DIM, qbuf, INNER, DIM);
    gemm_f16_f32<false><<<gkv, 256, 0, stream>>>(x16, DIM, Wkv, DIM, kvbuf, 2 * INNER, DIM);
    rope_l2norm_f16<<<NTOK * HEADS, 64, 0, stream>>>(qbuf, INNER, 0, q_scale + (size_t)l * DH, qh16);
    rope_l2norm_f16<<<NTOK * HEADS, 64, 0, stream>>>(kvbuf, 2 * INNER, 0, k_scale + (size_t)l * DH, kh16);
    v_transpose_f16<<<1024, 256, 0, stream>>>(kvbuf, vht16, (size_t)NTOK * INNER);
    flash_attn_kernel<<<gattn, 256, 0, stream>>>(qh16, kh16, vht16, o16);
    gemm_f16_f32<true><<<gq, 256, 0, stream>>>(o16, INNER, Wo, INNER, x, DIM, INNER);

    // feed-forward branch (SwiGLU fused into FF1 epilogue)
    ln_to_f16<<<NTOK, 256, 0, stream>>>(x, ff_g + (size_t)l * DIM, ff_b + (size_t)l * DIM, h16);
    gemm_ff1_swiglu<<<gff1, 256, 0, stream>>>(h16, W1, act16);
    gemm_f16_f32<true><<<gff2, 256, 0, stream>>>(act16, FF_INNER, W2, FF_INNER, x, DIM, FF_INNER);
  }
}
